// KDHR_42528766165502
// MI455X (gfx1250) — matrix operand order repaired and checked by reading, not compiled, not run on hardware
//
#include <hip/hip_runtime.h>
#include <hip/hip_bf16.h>

typedef __attribute__((ext_vector_type(16))) _Float16 v16h;
typedef __attribute__((ext_vector_type(8)))  float    v8f;

#define DEV static __device__ __forceinline__
#define USE_ASYNC_LDS 1

constexpr int N_SH = 1195, N_U = 805, N_I = 390, DD = 64, BB = 65536, E_SH = 100000;
constexpr int KPAD = 416;   // 13 * 32  (K padding for prescription GEMM, esT row length)
constexpr int NPAD = 816;   // 51 * 16  (N padding for eh GEMM)

// ---------------- workspace layout (bytes) ----------------
constexpr size_t ALN(size_t x) { return (x + 255) & ~size_t(255); }
constexpr size_t SZ_NODE = size_t(N_SH) * DD * 4;
constexpr size_t OFF_AGG1 = 0;
constexpr size_t OFF_AGG2 = ALN(OFF_AGG1 + SZ_NODE);
constexpr size_t OFF_CNT  = ALN(OFF_AGG2 + SZ_NODE);
constexpr size_t OFF_H1   = ALN(OFF_CNT + size_t(N_SH) * 4);
constexpr size_t OFF_H2   = ALN(OFF_H1 + SZ_NODE);
constexpr size_t OFF_SUM  = ALN(OFF_H2 + SZ_NODE);
constexpr size_t OFF_SQ    = OFF_SUM + 256;
constexpr size_t OFF_SCALE = OFF_SQ + 256;
constexpr size_t OFF_SHIFT = OFF_SCALE + 256;
constexpr size_t OFF_EST  = ALN(OFF_SHIFT + 256);                 // esT: 64 x 416 f16
constexpr size_t OFF_EHP  = ALN(OFF_EST + size_t(DD) * KPAD * 2); // eh: 816 x 64 f16
constexpr size_t OFF_WMH  = ALN(OFF_EHP + size_t(NPAD) * DD * 2); // Wm: 64 x 64 f16
constexpr size_t OFF_EBUF = ALN(OFF_WMH + size_t(DD) * DD * 2);   // e: 65536 x 64 f16
constexpr size_t ZERO_BYTES = OFF_EBUF;

// ---------------- async global->LDS copy (16B per lane, ASYNCcnt) ----------------
DEV void async_copy16(_Float16* ldst, const _Float16* gsrc) {
#if defined(__HIP_DEVICE_COMPILE__) && USE_ASYNC_LDS
  unsigned lo = (unsigned)(uintptr_t)ldst;                 // low 32 bits = LDS offset
  unsigned long long ga = (unsigned long long)(uintptr_t)gsrc;
  asm volatile("global_load_async_to_lds_b128 %0, %1, off"
               :: "v"(lo), "v"(ga) : "memory");
#else
  *(uint4*)ldst = *(const uint4*)gsrc;
#endif
}
DEV void wait_async0() {
#if defined(__HIP_DEVICE_COMPILE__) && USE_ASYNC_LDS
  asm volatile("s_wait_asynccnt 0" ::: "memory");
#endif
}

// ---------------- WMMA fragment helpers (contiguous k = 16*(lane/16) + s, same order for A and B) ----------------
// A: 16x32 f16; each K pair contiguous -> one b32 load per pair.
DEV v16h frag_a(const _Float16* Ab, int ld, int k0) {
  int lane = threadIdx.x & 31, m = lane & 15, hf = lane >> 4;
  union { v16h v; unsigned u[8]; } a;
#pragma unroll
  for (int p = 0; p < 8; ++p) {
    int kb = k0 + hf * 16 + 2 * p;
    a.u[p] = *(const unsigned*)&Ab[m * ld + kb];   // (m*ld+kb)*2 is 4B aligned
  }
  return a.v;
}
// B from an N-major source (S[n][k], i.e. B[k][n] = S[n][k]); pairs contiguous.
DEV v16h frag_bt(const _Float16* Sb, int ld, int k0, int n0) {
  int lane = threadIdx.x & 31, n = lane & 15, hf = lane >> 4;
  union { v16h v; unsigned u[8]; } b;
#pragma unroll
  for (int v = 0; v < 8; ++v) {
    int k = k0 + hf * 16 + 2 * v;
    b.u[v] = *(const unsigned*)&Sb[(size_t)(n0 + n) * ld + k];
  }
  return b.v;
}
DEV v8f wmma16(v16h a, v16h b, v8f c) {
  return __builtin_amdgcn_wmma_f32_16x16x32_f16(false, a, false, b, (short)0, c,
                                                false, false);
}

// ---------------- small kernels ----------------
__global__ void k_zero(uint4* p, int n4) {
  int i = blockIdx.x * 256 + threadIdx.x;
  if (i < n4) p[i] = uint4{0u, 0u, 0u, 0u};
}

__global__ void k_agg(const int* __restrict__ ei, const float* __restrict__ x,
                      float* __restrict__ agg, float* __restrict__ cnt) {
  int t = blockIdx.x * 256 + threadIdx.x;
  if (t >= E_SH * DD) return;
  int e = t >> 6, d = t & 63;
  int src = ei[e], dst = ei[E_SH + e];
  atomicAdd(&agg[dst * DD + d], x[src * DD + d]);
  if (cnt != nullptr && d == 0) atomicAdd(&cnt[dst], 1.0f);
}

__global__ void k_dense_tanh(const float* __restrict__ agg,
                             const float* __restrict__ cnt,
                             const float* __restrict__ W,
                             const float* __restrict__ b,
                             float* __restrict__ h) {
  int t = blockIdx.x * 256 + threadIdx.x;
  if (t >= N_SH * DD) return;
  int n = t >> 6, d = t & 63;
  const float* ar = agg + n * DD;
  const float* wr = W + d * DD;
  float s = 0.f;
#pragma unroll 8
  for (int k = 0; k < DD; ++k) s += ar[k] * wr[k];
  float c = fmaxf(cnt[n], 1.0f);
  h[t] = tanhf(s / c + b[d]);
}

__global__ void k_proj(const float* __restrict__ h2,
                       const float* __restrict__ Wp1, const float* __restrict__ bp1,
                       const float* __restrict__ Wp2, const float* __restrict__ bp2,
                       _Float16* __restrict__ esT, _Float16* __restrict__ ehp) {
  __shared__ float z[DD], tt[DD], red[DD];
  int node = blockIdx.x, d = threadIdx.x;
  z[d] = h2[node * DD + d];
  __syncthreads();
  float s = 0.f;
#pragma unroll 8
  for (int k = 0; k < DD; ++k) s += z[k] * Wp1[d * DD + k];
  s += bp1[d];
  tt[d] = (s > 0.f) ? s : (expf(s) - 1.f);   // ELU
  __syncthreads();
  float p = 0.f;
#pragma unroll 8
  for (int k = 0; k < DD; ++k) p += tt[k] * Wp2[d * DD + k];
  p += bp2[d];
  red[d] = p * p;
  __syncthreads();
  for (int st = 32; st > 0; st >>= 1) {
    if (d < st) red[d] += red[d + st];
    __syncthreads();
  }
  float pn = p * rsqrtf(red[0]);
  if (node < N_U) ehp[node * DD + d] = (_Float16)pn;          // eh: N-major rows
  else            esT[d * KPAD + (node - N_U)] = (_Float16)pn; // esT[n][k] (transposed)
}

__global__ void k_wmh(const float* __restrict__ Wm, _Float16* __restrict__ wmh) {
  int t = blockIdx.x * 256 + threadIdx.x;
  if (t < DD * DD) wmh[t] = (_Float16)Wm[t];   // Wm row-major == frag_bt source
}

__global__ void k_bnstats(const float* __restrict__ gsum, const float* __restrict__ gsq,
                          const float* __restrict__ gamma, const float* __restrict__ beta,
                          float* __restrict__ scale, float* __restrict__ shift) {
  int d = threadIdx.x;
  float mean = gsum[d] * (1.0f / BB);
  float var  = gsq[d] * (1.0f / BB) - mean * mean;
  float sc   = gamma[d] * rsqrtf(var + 1e-5f);
  scale[d] = sc;
  shift[d] = beta[d] - mean * sc;
}

// ---------------- GEMM 1: e = ((P @ es) / rowsum(P)) @ Wm^T + bm ----------------
__global__ __launch_bounds__(256) void k_gemm1(
    const float* __restrict__ P, const float* __restrict__ bm,
    const _Float16* __restrict__ esT, const _Float16* __restrict__ wmh,
    _Float16* __restrict__ ebuf, float* __restrict__ gsum, float* __restrict__ gsq) {
  __shared__ _Float16 btile[64 * 40];      // es^T chunk: 64 n-rows x 32 k (pad 40)
  __shared__ _Float16 ptile[128 * 40];     // P chunk (f16)
  __shared__ _Float16 wml[DD * DD];        // Wm (f16, row-major)
  __shared__ _Float16 ework[8 * 16 * 66];  // per-wave e_mid staging
  __shared__ float rows_s[128];
  __shared__ float csum[DD], csq[DD];

  const int tid = threadIdx.x;
  const int wv = tid >> 5, lane = tid & 31, m = lane & 15, hf = lane >> 4;
  const int rb = blockIdx.x * 128;
  const int myrow = rb + wv * 16;

  // stage Wm (f16) via async copy
  for (int i = tid; i < 512; i += 256)
    async_copy16(&wml[i * 8], &wmh[i * 8]);
  if (tid < DD) { csum[tid] = 0.f; csq[tid] = 0.f; }

  v8f acc[4] = {};
  float rowsum = 0.f;

  for (int ch = 0; ch < 13; ++ch) {
    __syncthreads();
    {  // stage es^T chunk: 64 rows x 32 halves = 256 lanes x 16B, async-to-LDS
      int n = tid >> 2, part = (tid & 3) * 8;
      async_copy16(&btile[n * 40 + part],
                   &esT[(size_t)n * KPAD + ch * 32 + part]);
    }
    if (ch < 12) {
      // full chunk: unconditional vectorized float2 stream (rows are 8B aligned)
      int r = tid >> 1, half = tid & 1;
      const float2* src =
          (const float2*)(P + (size_t)(rb + r) * N_I + ch * 32 + half * 16);
      float2 f[8];
#pragma unroll
      for (int j = 0; j < 8; ++j) f[j] = src[j];
      union { uint4 q[2]; _Float16 h[16]; } u;
#pragma unroll
      for (int j = 0; j < 8; ++j) {
        u.h[2 * j]     = (_Float16)f[j].x;
        u.h[2 * j + 1] = (_Float16)f[j].y;
      }
      uint4* dst = (uint4*)&ptile[r * 40 + half * 16];   // 16B aligned
      dst[0] = u.q[0];
      dst[1] = u.q[1];
    } else {
      // tail chunk (k 384..415, valid < 390): guarded scalar
      for (int i = tid; i < 128 * 32; i += 256) {
        int r = i >> 5, kk = i & 31, k = ch * 32 + kk;
        float v = (k < N_I) ? P[(size_t)(rb + r) * N_I + k] : 0.f;
        ptile[r * 40 + kk] = (_Float16)v;
      }
    }
    wait_async0();
    __syncthreads();

    if (hf == 0) {  // fused pre_sum: lanes 0..15 own one row; b128 LDS reads
      const uint4* pr4 = (const uint4*)&ptile[(wv * 16 + m) * 40];
#pragma unroll
      for (int j = 0; j < 4; ++j) {
        union { uint4 q; _Float16 h[8]; } u;
        u.q = pr4[j];
#pragma unroll
        for (int x = 0; x < 8; ++x) rowsum += (float)u.h[x];
      }
    }
    v16h a = frag_a(&ptile[wv * 16 * 40], 40, 0);
#pragma unroll
    for (int nt = 0; nt < 4; ++nt) {
      v16h b = frag_bt(btile, 40, 0, nt * 16);
      acc[nt] = wmma16(a, b, acc[nt]);
    }
  }
  __syncthreads();
  if (hf == 0) rows_s[wv * 16 + m] = rowsum;
  __syncthreads();

  // divide by pre_sum, stage as f16 A-operand for the 64x64 epilogue
#pragma unroll
  for (int nt = 0; nt < 4; ++nt) {
#pragma unroll
    for (int r = 0; r < 8; ++r) {
      int row = r + hf * 8;
      float inv = 1.f / rows_s[wv * 16 + row];
      ework[(wv * 16 + row) * 66 + nt * 16 + m] = (_Float16)(acc[nt][r] * inv);
    }
  }
  __syncthreads();

  v16h a0 = frag_a(&ework[wv * 16 * 66], 66, 0);
  v16h a1 = frag_a(&ework[wv * 16 * 66], 66, 32);
  v8f acc2[4] = {};
#pragma unroll
  for (int nt = 0; nt < 4; ++nt) {
    v16h b0 = frag_bt(wml, DD, 0, nt * 16);   // B[k][n] = Wm[n][k]
    acc2[nt] = wmma16(a0, b0, acc2[nt]);
    v16h b1 = frag_bt(wml, DD, 32, nt * 16);
    acc2[nt] = wmma16(a1, b1, acc2[nt]);
  }

  // + bm, store e (f16), accumulate BN statistics
#pragma unroll
  for (int nt = 0; nt < 4; ++nt) {
    int col = nt * 16 + m;
    float bmv = bm[col];
    float ps = 0.f, pq = 0.f;
#pragma unroll
    for (int r = 0; r < 8; ++r) {
      float v = acc2[nt][r] + bmv;
      ebuf[(size_t)(myrow + r + hf * 8) * DD + col] = (_Float16)v;
      ps += v; pq += v * v;
    }
    ps += __shfl_xor(ps, 16);
    pq += __shfl_xor(pq, 16);
    if (hf == 0) { atomicAdd(&csum[col], ps); atomicAdd(&csq[col], pq); }
  }
  __syncthreads();
  if (tid < DD) { atomicAdd(&gsum[tid], csum[tid]); atomicAdd(&gsq[tid], csq[tid]); }
}

// ---------------- GEMM 2: out = relu(BN(e)) @ eh^T ----------------
__global__ __launch_bounds__(256) void k_gemm2(
    const _Float16* __restrict__ ebuf, const _Float16* __restrict__ ehp,
    const float* __restrict__ scale, const float* __restrict__ shift,
    float* __restrict__ out) {
  __shared__ _Float16 el[16 * 72];
  const int tid = threadIdx.x;
  const int wv = tid >> 5, lane = tid & 31, m = lane & 15, hf = lane >> 4;
  const int rb = blockIdx.x * 16;

  {  // BN + ReLU applied while staging A; 4 halves per thread, vector ld/st
    int i = tid * 4, r = i >> 6, c = i & 63;
    union { uint2 q; _Float16 h[4]; } u;
    u.q = *(const uint2*)&ebuf[(size_t)(rb + r) * DD + c];
#pragma unroll
    for (int j = 0; j < 4; ++j) {
      float v = (float)u.h[j];
      u.h[j] = (_Float16)fmaxf(v * scale[c + j] + shift[c + j], 0.f);
    }
    *(uint2*)&el[r * 72 + c] = u.q;
  }
  __syncthreads();

  v16h a0 = frag_a(el, 72, 0);
  v16h a1 = frag_a(el, 72, 32);

  for (int nt = wv; nt < NPAD / 16; nt += 8) {
    int n0 = nt * 16;
    v16h b0 = frag_bt(ehp, DD, 0, n0);   // L2-resident eh (f16, N-padded)
    v16h b1 = frag_bt(ehp, DD, 32, n0);
    v8f c = {};
    c = wmma16(a0, b0, c);
    c = wmma16(a1, b1, c);
    int u = n0 + m;
    if (u < N_U) {
#pragma unroll
      for (int r = 0; r < 8; ++r)
        out[(size_t)(rb + r + hf * 8) * N_U + u] = c[r];
    }
  }
}

// ---------------- host launcher ----------------
extern "C" void kernel_launch(void* const* d_in, const int* in_sizes, int n_in,
                              void* d_out, int out_size, void* d_ws, size_t ws_size,
                              hipStream_t stream) {
  (void)in_sizes; (void)n_in; (void)out_size; (void)ws_size;
  const int*   eSH  = (const int*)d_in[0];
  const float* P    = (const float*)d_in[3];
  const float* sh   = (const float*)d_in[4];
  const float* W1   = (const float*)d_in[5];
  const float* b1   = (const float*)d_in[6];
  const float* W2   = (const float*)d_in[7];
  const float* b2   = (const float*)d_in[8];
  const float* Wp1  = (const float*)d_in[9];
  const float* bp1  = (const float*)d_in[10];
  const float* Wp2  = (const float*)d_in[11];
  const float* bp2  = (const float*)d_in[12];
  const float* Wm   = (const float*)d_in[13];
  const float* bm   = (const float*)d_in[14];
  const float* gam  = (const float*)d_in[15];
  const float* bet  = (const float*)d_in[16];

  char* w = (char*)d_ws;
  float*     agg1  = (float*)(w + OFF_AGG1);
  float*     agg2  = (float*)(w + OFF_AGG2);
  float*     cnt   = (float*)(w + OFF_CNT);
  float*     h1    = (float*)(w + OFF_H1);
  float*     h2    = (float*)(w + OFF_H2);
  float*     gsum  = (float*)(w + OFF_SUM);
  float*     gsq   = (float*)(w + OFF_SQ);
  float*     scale = (float*)(w + OFF_SCALE);
  float*     shift = (float*)(w + OFF_SHIFT);
  _Float16*  esT   = (_Float16*)(w + OFF_EST);
  _Float16*  ehp   = (_Float16*)(w + OFF_EHP);
  _Float16*  wmh   = (_Float16*)(w + OFF_WMH);
  _Float16*  ebuf  = (_Float16*)(w + OFF_EBUF);

  const int nz4 = (int)(ZERO_BYTES / 16);
  k_zero<<<(nz4 + 255) / 256, 256, 0, stream>>>((uint4*)w, nz4);

  // GCN layer 1 + 2 (aggregate-then-transform, equivalent to per-edge x@W.T)
  k_agg<<<(E_SH * DD) / 256, 256, 0, stream>>>(eSH, sh, agg1, cnt);
  k_dense_tanh<<<(N_SH * DD + 255) / 256, 256, 0, stream>>>(agg1, cnt, W1, b1, h1);
  k_agg<<<(E_SH * DD) / 256, 256, 0, stream>>>(eSH, h1, agg2, nullptr);
  k_dense_tanh<<<(N_SH * DD + 255) / 256, 256, 0, stream>>>(agg2, cnt, W2, b2, h2);

  // projections -> esT (64 x 416, transposed) / eh (816 x 64), f16
  k_wmh<<<16, 256, 0, stream>>>(Wm, wmh);
  k_proj<<<N_SH, DD, 0, stream>>>(h2, Wp1, bp1, Wp2, bp2, esT, ehp);

  // fused: P@es, /pre_sum, @Wm^T + bm, BN stats, e -> f16
  k_gemm1<<<BB / 128, 256, 0, stream>>>(P, bm, esT, wmh, ebuf, gsum, gsq);
  k_bnstats<<<1, DD, 0, stream>>>(gsum, gsq, gam, bet, scale, shift);

  // out = relu(BN(e)) @ eh^T
  k_gemm2<<<BB / 16, 256, 0, stream>>>(ebuf, ehp, scale, shift, (float*)d_out);
}
